// DGCNN_Grouper_11897059410080
// MI455X (gfx1250) — matrix operand order repaired and checked by reading, not compiled, not run on hardware
//
#include <hip/hip_runtime.h>

typedef __attribute__((ext_vector_type(16))) _Float16 v16h;
typedef __attribute__((ext_vector_type(8)))  float    v8f;

#define LRELU_SLOPE 0.2f
#define GN_EPS 1e-5f
#define QT 8  // queries per wave in edge_mm

__device__ __forceinline__ v8f wmma16x16x32(v16h a, v16h b, v8f c) {
  // D = A(16x32 f16) * B(32x16 f16) + C(16x16 f32)
  return __builtin_amdgcn_wmma_f32_16x16x32_f16(false, a, false, b, (short)0, c,
                                                false, false);
}

// Async global->LDS copy (gfx1250), tracked by ASYNCcnt.
__device__ __forceinline__ void async_load_f32(unsigned lds_off,
                                               const float* gptr) {
  asm volatile("global_load_async_to_lds_b32 %0, %1, off"
               :
               : "v"(lds_off), "v"(gptr)
               : "memory");
}
__device__ __forceinline__ void async_wait_2() {
  asm volatile("s_wait_asynccnt 0x2" ::: "memory");
}
__device__ __forceinline__ void async_wait_0() {
  asm volatile("s_wait_asynccnt 0x0" ::: "memory");
}
__device__ __forceinline__ unsigned lds_off_of(const void* p) {
  return (unsigned)(size_t)p;  // low 32 bits of generic LDS address = DS offset
}

// ---------------------------------------------------------------------------
// Stage 0: f = w_in @ x + b_in   (B,3,N) -> (B,8,N); also emits inpc_f output
// ---------------------------------------------------------------------------
__global__ void proj_in_kernel(const float* __restrict__ x,
                               const float* __restrict__ w,
                               const float* __restrict__ bias,
                               float* __restrict__ f,
                               float* __restrict__ inpc,
                               int B_, int N_) {
  int t = blockIdx.x * blockDim.x + threadIdx.x;
  if (t >= B_ * N_) return;
  int b = t / N_, n = t % N_;
  const float* xb = x + (size_t)b * 3 * N_;
  float x0 = xb[n], x1 = xb[N_ + n], x2 = xb[2 * N_ + n];
#pragma unroll
  for (int o = 0; o < 8; ++o) {
    float v = w[o * 3 + 0] * x0 + w[o * 3 + 1] * x1 + w[o * 3 + 2] * x2 + bias[o];
    f[((size_t)b * 8 + o) * N_ + n] = v;
    inpc[((size_t)b * 8 + o) * N_ + n] = v;
  }
}

// ---------------------------------------------------------------------------
// kNN (k=16): one wave per 16 queries. Key tiles stream through LDS via
// double-buffered async global->LDS loads; WMMA computes q.k tiles; lanes
// 0..15 maintain per-query sorted top-16 with early reject.
// ---------------------------------------------------------------------------
__global__ void __launch_bounds__(32)
knn_kernel(const float* __restrict__ cq, int nq,
           const float* __restrict__ ck, int nk,
           int* __restrict__ idxOut /* [B][nq][16] */) {
  __shared__ float sqq[16];
  __shared__ float ktile[2][48];  // 3 coord rows x 16 keys, double buffered
  __shared__ float stile[16 * 16];
  const int l  = threadIdx.x;
  const int hi = l >> 4;
  const int lo = l & 15;
  const int b  = blockIdx.y;
  const int qBase = blockIdx.x * 16;
  const float* cqb = cq + (size_t)b * 3 * nq;
  const float* ckb = ck + (size_t)b * 3 * nk;

  // A fragment: 16 queries (M) x 32 (K = coord dim, zero padded)
  v16h a;
#pragma unroll
  for (int h = 0; h < 16; ++h) {
    int kd = (h < 8) ? (h + 8 * hi) : (h + 8 + 8 * hi);
    float v = (kd < 3) ? cqb[kd * nq + qBase + lo] : 0.0f;
    a[h] = (_Float16)v;
  }
  if (l < 16) {
    float q0 = cqb[0 * nq + qBase + l];
    float q1 = cqb[1 * nq + qBase + l];
    float q2 = cqb[2 * nq + qBase + l];
    sqq[l] = q0 * q0 + q1 * q1 + q2 * q2;
  }
  __syncthreads();

  // Prime the pipeline: async-load key tile 0 (rows 0,1 by all lanes; row 2
  // by lanes 0..15).
  async_load_f32(lds_off_of(&ktile[0][l]), ckb + hi * nk + lo);
  if (l < 16)
    async_load_f32(lds_off_of(&ktile[0][32 + l]), ckb + 2 * nk + l);

  float bd[16];
  int   bi[16];
#pragma unroll
  for (int j = 0; j < 16; ++j) { bd[j] = 3.4e38f; bi[j] = 0; }

  const int ntiles = nk / 16;
  for (int t = 0; t < ntiles; ++t) {
    const int buf = t & 1;
    const int kBase = t * 16;
    if (t + 1 < ntiles) {
      const int kB2 = kBase + 16;
      __builtin_prefetch(ckb + kB2 + 16, 0, 1);
      async_load_f32(lds_off_of(&ktile[buf ^ 1][l]), ckb + hi * nk + kB2 + lo);
      if (l < 16)
        async_load_f32(lds_off_of(&ktile[buf ^ 1][32 + l]), ckb + 2 * nk + kB2 + l);
      async_wait_2();  // only the tile just issued may remain in flight
    } else {
      async_wait_0();
    }

    // B fragment: 32 (K = coord dim padded) x 16 keys (N) from LDS
    v16h bf;
#pragma unroll
    for (int h = 0; h < 16; ++h) {
      int kd = (h < 8) ? (h + 8 * hi) : (h + 8 + 8 * hi);
      float v = (kd < 3) ? ktile[buf][kd * 16 + lo] : 0.0f;
      bf[h] = (_Float16)v;
    }
    float kk = 0.0f;
#pragma unroll
    for (int kd = 0; kd < 3; ++kd) {
      float v = ktile[buf][kd * 16 + lo];
      kk += v * v;
    }

    v8f acc = {};
    acc = wmma16x16x32(a, bf, acc);  // acc[M=r+8*hi][N=lo] = q.k

#pragma unroll
    for (int r = 0; r < 8; ++r) {
      int m = r + 8 * hi;
      stile[m * 16 + lo] = sqq[m] + kk - 2.0f * acc[r];
    }
    __syncthreads();

    if (l < 16) {
#pragma unroll
      for (int j = 0; j < 16; ++j) {
        float cd = stile[l * 16 + j];
        if (cd < bd[15]) {  // early reject: almost always false after warm-up
          int ci = kBase + j;
#pragma unroll
          for (int s = 0; s < 16; ++s) {
            if (cd < bd[s]) {
              float td = bd[s]; bd[s] = cd; cd = td;
              int   ti = bi[s]; bi[s] = ci; ci = ti;
            }
          }
        }
      }
    }
    __syncthreads();
  }

  if (l < 16) {
    int* o = idxOut + ((size_t)b * nq + qBase + l) * 16;
#pragma unroll
    for (int j = 0; j < 16; ++j) o[j] = bi[j];
  }
}

// ---------------------------------------------------------------------------
// Edge conv GEMM: h[b,o,q,k] = sum_c w[o,c] * g[c,q,k],
//   g = concat(fk[:,idx]-fq, fq) built on the fly in the B fragment.
// One wave per (8-query chunk, 16-output tile, batch): weight A-fragments
// are built once and reused across the 8 queries. Also accumulates the
// GroupNorm sum / sumsq via f32 global atomics (one pair per lane).
// ---------------------------------------------------------------------------
__global__ void __launch_bounds__(32)
edge_mm_kernel(const float* __restrict__ fq, int nq,
               const float* __restrict__ fk, int nk,
               const int*   __restrict__ nbr /* [B][nq][16] */,
               const float* __restrict__ w   /* [O][2C] */,
               float* __restrict__ hOut      /* [B][O][nq][16] */,
               float* __restrict__ stats     /* [B][4][2] */,
               int C, int O) {
  const int l  = threadIdx.x;
  const int hi = l >> 4;
  const int lo = l & 15;
  const int qBase = blockIdx.x * QT;
  const int oBase = blockIdx.y * 16;
  const int b     = blockIdx.z;
  const int twoC  = 2 * C;
  const int chunks = (twoC + 31) / 32;
  const float* fqb = fq + (size_t)b * C * nq;
  const float* fkb = fk + (size_t)b * C * nk;

  // Weight A-fragments, invariant across queries (<= 4 chunks -> 32 VGPRs)
  v16h afr[4];
#pragma unroll
  for (int c0 = 0; c0 < 4; ++c0) {
    if (c0 < chunks) {
#pragma unroll
      for (int h = 0; h < 16; ++h) {
        int ka = c0 * 32 + ((h < 8) ? (h + 8 * hi) : (h + 8 + 8 * hi));
        afr[c0][h] = (ka < twoC) ? (_Float16)w[(size_t)(oBase + lo) * twoC + ka]
                                 : (_Float16)0.0f;
      }
    }
  }

  const int gsz = O >> 2;                  // channels per group (>=8)
  const int g0  = (oBase + 8 * hi) / gsz;  // all 8 lane values share a group
  float s = 0.f, ss = 0.f;

  for (int qq = 0; qq < QT; ++qq) {
    const int q = qBase + qq;
    const int nIdx = nbr[((size_t)b * nq + q) * 16 + lo];

    v8f acc = {};
#pragma unroll
    for (int c0 = 0; c0 < 4; ++c0) {
      if (c0 < chunks) {
        v16h bf;
#pragma unroll
        for (int h = 0; h < 16; ++h) {
          int kb = c0 * 32 + ((h < 8) ? (h + 8 * hi) : (h + 8 + 8 * hi));
          float v = 0.0f;
          if (kb < C)          v = fkb[(size_t)kb * nk + nIdx] - fqb[(size_t)kb * nq + q];
          else if (kb < twoC)  v = fqb[(size_t)(kb - C) * nq + q];
          bf[h] = (_Float16)v;
        }
        acc = wmma16x16x32(afr[c0], bf, acc);
      }
    }
#pragma unroll
    for (int r = 0; r < 8; ++r) {
      int o = oBase + 8 * hi + r;
      float v = acc[r];
      hOut[(((size_t)b * O + o) * nq + q) * 16 + lo] = v;
      s += v; ss += v * v;
    }
  }
  atomicAdd(&stats[((size_t)b * 4 + g0) * 2 + 0], s);
  atomicAdd(&stats[((size_t)b * 4 + g0) * 2 + 1], ss);
}

// ---------------------------------------------------------------------------
// GroupNorm + LeakyReLU + max over the 16 neighbors
// ---------------------------------------------------------------------------
__global__ void gn_max_kernel(const float* __restrict__ hIn,
                              const float* __restrict__ stats,
                              const float* __restrict__ gw,
                              const float* __restrict__ gb,
                              float* __restrict__ out /* [B][O][nq] */,
                              int B_, int O, int nq) {
  int t = blockIdx.x * blockDim.x + threadIdx.x;
  if (t >= B_ * O * nq) return;
  int q = t % nq;
  int o = (t / nq) % O;
  int b = t / (nq * O);
  int gsz = O >> 2;
  int g = o / gsz;
  float cnt  = (float)gsz * (float)nq * 16.0f;
  float mean = stats[((size_t)b * 4 + g) * 2 + 0] / cnt;
  float var  = stats[((size_t)b * 4 + g) * 2 + 1] / cnt - mean * mean;
  float inv  = rsqrtf(var + GN_EPS);
  float sc = gw[o], bi = gb[o];
  const float* hp = hIn + (((size_t)b * O + o) * nq + q) * 16;
  float m = -3.4e38f;
#pragma unroll
  for (int k = 0; k < 16; ++k) {
    float v = (hp[k] - mean) * inv * sc + bi;
    v = (v < 0.f) ? v * LRELU_SLOPE : v;
    m = fmaxf(m, v);
  }
  out[((size_t)b * O + o) * nq + q] = m;
}

// ---------------------------------------------------------------------------
// Farthest point sampling: one 256-thread workgroup per batch,
// running min-distance array in LDS, tree argmax per iteration.
// ---------------------------------------------------------------------------
__global__ void __launch_bounds__(256)
fps_kernel(const float* __restrict__ coor /* [B][3][n] */, int n, int m,
           int* __restrict__ idxOut /* [B][m] */) {
  __shared__ float sd[4096];
  __shared__ float rv[256];
  __shared__ int   ri[256];
  __shared__ int   sfar;
  const int tid = threadIdx.x;
  const int b = blockIdx.x;
  const float* cb = coor + (size_t)b * 3 * n;
  for (int i = tid; i < n; i += 256) sd[i] = 1e10f;
  if (tid == 0) sfar = 0;
  __syncthreads();
  for (int it = 0; it < m; ++it) {
    int far = sfar;
    if (tid == 0) idxOut[(size_t)b * m + it] = far;
    float cx = cb[far], cy = cb[n + far], cz = cb[2 * n + far];
    float lmax = -1.f; int limax = 0;
    for (int i = tid; i < n; i += 256) {
      float dx = cb[i] - cx, dy = cb[n + i] - cy, dz = cb[2 * n + i] - cz;
      float d = dx * dx + dy * dy + dz * dz;
      float cur = fminf(sd[i], d);
      sd[i] = cur;
      if (cur > lmax) { lmax = cur; limax = i; }
    }
    rv[tid] = lmax; ri[tid] = limax;
    __syncthreads();
    for (int s = 128; s > 0; s >>= 1) {
      if (tid < s && rv[tid + s] > rv[tid]) {
        rv[tid] = rv[tid + s]; ri[tid] = ri[tid + s];
      }
      __syncthreads();
    }
    if (tid == 0) sfar = ri[0];
    __syncthreads();
  }
}

// ---------------------------------------------------------------------------
// Gather coords + features at FPS indices
// ---------------------------------------------------------------------------
__global__ void gather_kernel(const float* __restrict__ coor,
                              const float* __restrict__ f,
                              const int* __restrict__ idx,
                              float* __restrict__ coorOut,
                              float* __restrict__ fOut,
                              int C, int n, int m, int B_) {
  int t = blockIdx.x * blockDim.x + threadIdx.x;
  if (t >= B_ * m) return;
  int j = t % m, b = t / m;
  int i = idx[(size_t)b * m + j];
  for (int c = 0; c < 3; ++c)
    coorOut[((size_t)b * 3 + c) * m + j] = coor[((size_t)b * 3 + c) * n + i];
  for (int c = 0; c < C; ++c)
    fOut[((size_t)b * C + c) * m + j] = f[((size_t)b * C + c) * n + i];
}

__global__ void zero_kernel(float* p, int n) {
  int t = blockIdx.x * blockDim.x + threadIdx.x;
  if (t < n) p[t] = 0.f;
}

// ---------------------------------------------------------------------------
extern "C" void kernel_launch(void* const* d_in, const int* in_sizes, int n_in,
                              void* d_out, int out_size, void* d_ws, size_t ws_size,
                              hipStream_t stream) {
  (void)in_sizes; (void)n_in; (void)out_size; (void)ws_size;
  const int B_ = 4, N_ = 4096, M1 = 2048, M2 = 1024;

  const float* x    = (const float*)d_in[0];
  const float* w_in = (const float*)d_in[1];
  const float* b_in = (const float*)d_in[2];
  const float* w1   = (const float*)d_in[3];
  const float* g1w  = (const float*)d_in[4];
  const float* g1b  = (const float*)d_in[5];
  const float* w2   = (const float*)d_in[6];
  const float* g2w  = (const float*)d_in[7];
  const float* g2b  = (const float*)d_in[8];
  const float* w3   = (const float*)d_in[9];
  const float* g3w  = (const float*)d_in[10];
  const float* g3b  = (const float*)d_in[11];
  const float* w4   = (const float*)d_in[12];
  const float* g4w  = (const float*)d_in[13];
  const float* g4b  = (const float*)d_in[14];

  // d_out: coor(4*3*1024) | f(4*128*1024) | xyz1(4*3*2048) | point1(4*64*2048)
  //        | inpc_f(4*8*4096)
  float* out    = (float*)d_out;
  float* o_coor = out;
  float* o_f    = out + 12288;
  float* o_xyz1 = out + 536576;
  float* o_pt1  = out + 561152;
  float* o_inpc = out + 1085440;

  float* ws    = (float*)d_ws;
  float* f0    = ws;                 // B*8*N    = 131072
  float* f1    = f0 + 131072;        // B*32*N   = 524288
  float* f3    = f1 + 524288;        // B*64*M1  = 524288
  float* fqbuf = f3 + 524288;        // downsampled query feats, 262144
  float* hbuf  = fqbuf + 262144;     // B*O*Nq*16 = 8388608 (same all stages)
  float* stats = hbuf + 8388608;     // 32
  int*   idxb  = (int*)(stats + 32); // B*N*16 = 262144 ints
  int*   fpsi  = idxb + 262144;      // B*M1   = 8192 ints

  // stage 0: input projection (also inpc_f output)
  proj_in_kernel<<<(B_ * N_ + 255) / 256, 256, 0, stream>>>(x, w_in, b_in, f0,
                                                            o_inpc, B_, N_);
  // stage 1: knn(4096 vs 4096), edge conv 16->32
  knn_kernel<<<dim3(N_ / 16, B_), 32, 0, stream>>>(x, N_, x, N_, idxb);
  zero_kernel<<<1, 32, 0, stream>>>(stats, 32);
  edge_mm_kernel<<<dim3(N_ / QT, 2, B_), 32, 0, stream>>>(f0, N_, f0, N_, idxb,
                                                          w1, hbuf, stats, 8, 32);
  gn_max_kernel<<<(B_ * 32 * N_ + 255) / 256, 256, 0, stream>>>(
      hbuf, stats, g1w, g1b, f1, B_, 32, N_);

  // FPS 4096 -> 2048, gather coords (xyz1 output) + feats
  fps_kernel<<<B_, 256, 0, stream>>>(x, N_, M1, fpsi);
  gather_kernel<<<(B_ * M1 + 255) / 256, 256, 0, stream>>>(
      x, f1, fpsi, o_xyz1, fqbuf, 32, N_, M1, B_);

  // stage 2: knn(2048 vs 4096), edge conv 64->64 (point1 output)
  knn_kernel<<<dim3(M1 / 16, B_), 32, 0, stream>>>(o_xyz1, M1, x, N_, idxb);
  zero_kernel<<<1, 32, 0, stream>>>(stats, 32);
  edge_mm_kernel<<<dim3(M1 / QT, 4, B_), 32, 0, stream>>>(
      fqbuf, M1, f1, N_, idxb, w2, hbuf, stats, 32, 64);
  gn_max_kernel<<<(B_ * 64 * M1 + 255) / 256, 256, 0, stream>>>(
      hbuf, stats, g2w, g2b, o_pt1, B_, 64, M1);

  // stage 3: knn(2048 vs 2048), edge conv 128->64
  knn_kernel<<<dim3(M1 / 16, B_), 32, 0, stream>>>(o_xyz1, M1, o_xyz1, M1, idxb);
  zero_kernel<<<1, 32, 0, stream>>>(stats, 32);
  edge_mm_kernel<<<dim3(M1 / QT, 4, B_), 32, 0, stream>>>(
      o_pt1, M1, o_pt1, M1, idxb, w3, hbuf, stats, 64, 64);
  gn_max_kernel<<<(B_ * 64 * M1 + 255) / 256, 256, 0, stream>>>(
      hbuf, stats, g3w, g3b, f3, B_, 64, M1);

  // FPS 2048 -> 1024, gather coords (coor output) + feats
  fps_kernel<<<B_, 256, 0, stream>>>(o_xyz1, M1, M2, fpsi);
  gather_kernel<<<(B_ * M2 + 255) / 256, 256, 0, stream>>>(
      o_xyz1, f3, fpsi, o_coor, fqbuf, 64, M1, M2, B_);

  // stage 4: knn(1024 vs 2048), edge conv 128->128 (f output)
  knn_kernel<<<dim3(M2 / 16, B_), 32, 0, stream>>>(o_coor, M2, o_xyz1, M1, idxb);
  zero_kernel<<<1, 32, 0, stream>>>(stats, 32);
  edge_mm_kernel<<<dim3(M2 / QT, 8, B_), 32, 0, stream>>>(
      fqbuf, M2, f3, M1, idxb, w4, hbuf, stats, 64, 128);
  gn_max_kernel<<<(B_ * 128 * M2 + 255) / 256, 256, 0, stream>>>(
      hbuf, stats, g4w, g4b, o_f, B_, 128, M2);
}
